// SelectiveSSM_33440615367094
// MI455X (gfx1250) — matrix operand order repaired and checked by reading, not compiled, not run on hardware
//
#include <hip/hip_runtime.h>
#include <hip/hip_bf16.h>
#include <math.h>

#define Bsz 2
#define Lsz 2048
#define Dsz 1024
#define Nsz 16
#define DTC 0.1f

typedef __attribute__((ext_vector_type(16))) __bf16 v16bf;
typedef __attribute__((ext_vector_type(8)))  float  v8f;
typedef __attribute__((ext_vector_type(4)))  int    v4i;

// ---- CDNA5 async global->LDS path (guarded; falls back to plain LDS stores) ----
#if defined(__has_builtin)
#if __has_builtin(__builtin_amdgcn_global_load_async_to_lds_b128)
#define USE_ASYNC 1
#endif
#endif

#if defined(USE_ASYNC)
#if __has_builtin(__builtin_amdgcn_s_wait_asynccnt)
#define WAIT_ASYNC(n) __builtin_amdgcn_s_wait_asynccnt(n)
#else
#define WAIT_ASYNC(n) asm volatile("s_wait_asynccnt %0" ::"n"(n) : "memory")
#endif
#else
#define WAIT_ASYNC(n) ((void)0)
#endif

__device__ __forceinline__ void async_copy_b128(const float* g, float* l) {
#if defined(USE_ASYNC)
    __builtin_amdgcn_global_load_async_to_lds_b128(
        (__attribute__((address_space(1))) v4i*)g,
        (__attribute__((address_space(3))) v4i*)l, 0, 0);
#else
    *(float4*)l = *(const float4*)g;
#endif
}

// ---------------- Kernel 1: gc = mean_L(x)  [B,D] ----------------
__global__ __launch_bounds__(256) void k_gc(const float* __restrict__ x,
                                            float* __restrict__ gc) {
    __shared__ float sm[256];
    int t  = threadIdx.x;
    int lp = t & 31;
    int ch = t >> 5;
    int pair = blockIdx.x * 32 + lp;
    int b = pair / Dsz, d = pair % Dsz;
    const float* px = x + ((size_t)b * Lsz + (size_t)ch * 256) * Dsz + d;
    float s = 0.f;
    for (int i = 0; i < 256; ++i) s += px[(size_t)i * Dsz];
    sm[t] = s;
    __syncthreads();
    if (ch == 0) {
        float tot = 0.f;
        #pragma unroll
        for (int c = 0; c < 8; ++c) tot += sm[lp + 32 * c];
        gc[pair] = tot * (1.0f / (float)Lsz);
    }
}

// ---------------- Kernel 2: liquid time-constant scaler -> sel [B,D] ----------------
__global__ __launch_bounds__(256) void k_liquid(
    const float* __restrict__ gc,
    const float* __restrict__ Wi_W, const float* __restrict__ Wi_b,
    const float* __restrict__ Wr_W,
    const float* __restrict__ sel_W, const float* __restrict__ sel_b,
    const float* __restrict__ tau, const float* __restrict__ h0,
    float* __restrict__ sel) {
    __shared__ float nh[Bsz * Nsz];
    int t = threadIdx.x;
    if (t < Bsz * Nsz) {
        int b = t >> 4, n = t & 15;
        float inp = Wi_b[n];
        const float* g = gc + (size_t)b * Dsz;
        const float* w = Wi_W + (size_t)n * Dsz;
        for (int k = 0; k < Dsz; ++k) inp += g[k] * w[k];
        float cur = h0[n];
        float rec = 0.f;
        #pragma unroll
        for (int m = 0; m < Nsz; ++m) rec += h0[m] * Wr_W[n * Nsz + m];
        float tc = fminf(fmaxf(tau[n], 0.1f), 10.f);
        float dh = (-cur + tanhf(inp + rec)) / tc;
        nh[t] = cur + DTC * dh;
    }
    __syncthreads();
    for (int idx = t; idx < Bsz * Dsz; idx += blockDim.x) {
        int b = idx / Dsz, d = idx % Dsz;
        float s = sel_b[d];
        #pragma unroll
        for (int n = 0; n < Nsz; ++n) s += nh[b * Nsz + n] * sel_W[d * Nsz + n];
        sel[idx] = 1.f / (1.f + __expf(-s));
    }
}

// ---------------- Kernel 3: LDS-staged WMMA GEMM ----------------
// Block = one 16-row tile of xs; 8 waves; wave w computes column tiles w*8..w*8+7
// of delta (+ wave6: Bm tile, wave7: Cm tile). xs slab double-buffered in LDS via
// async global->LDS copies; sel gate folded into A-fragment.
__global__ __launch_bounds__(256) void k_gemm(
    const float* __restrict__ x, const float* __restrict__ sel,
    const float* __restrict__ delta_W, const float* __restrict__ delta_b,
    const float* __restrict__ dt_bias,
    const float* __restrict__ B_W, const float* __restrict__ B_b,
    const float* __restrict__ C_W, const float* __restrict__ C_b,
    float* __restrict__ delta, float* __restrict__ Bm, float* __restrict__ Cm) {
    __shared__ float xslab[2][16 * 32];   // 2 x 2KB (f32 xs slab, double-buffered)
    __shared__ float selS[Dsz];           // 4KB (sel row for this batch)

    const int NK = Dsz / 32;              // 32 K-steps
    int tid  = threadIdx.x;
    int w    = tid >> 5;                  // wave id 0..7
    int lane = tid & 31;
    int hf   = lane >> 4;                 // K-half
    int n    = lane & 15;                 // A row / B col / D col

    int rt = blockIdx.x;                  // row tile (16 rows of b*L+l)
    int b  = (rt * 16) / Lsz;

    // stage sel row (1024 floats, one float4 per thread)
    *(float4*)&selS[tid * 4] = *(const float4*)(sel + (size_t)b * Dsz + tid * 4);

    // weight row base for this wave's 8 delta tiles: row = (w*8+j)*16 + n
    const float* wbase = delta_W + (size_t)(w * 128 + n) * Dsz;
    const float* wextra = (w == 6) ? (B_W + (size_t)n * Dsz)
                        : (w == 7) ? (C_W + (size_t)n * Dsz) : nullptr;

    // prologue: stage slab 0 (waves 0..3, one b128 per lane)
    if (tid < 128) {
        const float* src = x + (size_t)(rt * 16 + (tid >> 3)) * Dsz + ((tid & 7) * 4);
        async_copy_b128(src, &xslab[0][tid * 4]);
    }

    v8f a0 = {}, a1 = {}, a2 = {}, a3 = {}, a4 = {}, a5 = {}, a6 = {}, a7 = {}, a8 = {};

    for (int ks = 0; ks < NK; ++ks) {
        int kb  = ks * 32;
        int buf = ks & 1;
        if (ks + 1 < NK) {
            if (tid < 128) {
                const float* src = x + (size_t)(rt * 16 + (tid >> 3)) * Dsz +
                                   (kb + 32 + (tid & 7) * 4);
                async_copy_b128(src, &xslab[buf ^ 1][tid * 4]);
            }
            WAIT_ASYNC(1);
        } else {
            WAIT_ASYNC(0);
        }
        __syncthreads();   // slab[buf] (and selS on ks==0) visible to all waves

        // Build A fragment: element e -> k = hf*16 + e, row = n
        v16bf afr;
        #pragma unroll
        for (int g = 0; g < 2; ++g) {
            int k0 = hf * 16 + g * 8;
            float4 x0 = *(const float4*)&xslab[buf][n * 32 + k0];
            float4 x1 = *(const float4*)&xslab[buf][n * 32 + k0 + 4];
            float4 s0 = *(const float4*)&selS[kb + k0];
            float4 s1 = *(const float4*)&selS[kb + k0 + 4];
            afr[g * 8 + 0] = (__bf16)(x0.x * s0.x);
            afr[g * 8 + 1] = (__bf16)(x0.y * s0.y);
            afr[g * 8 + 2] = (__bf16)(x0.z * s0.z);
            afr[g * 8 + 3] = (__bf16)(x0.w * s0.w);
            afr[g * 8 + 4] = (__bf16)(x1.x * s1.x);
            afr[g * 8 + 5] = (__bf16)(x1.y * s1.y);
            afr[g * 8 + 6] = (__bf16)(x1.z * s1.z);
            afr[g * 8 + 7] = (__bf16)(x1.w * s1.w);
        }

        // 8 delta tiles + possible extra tile: B element e -> k = hf*16 + e
        #define TILE_STEP(ACC, WROW)                                              \
        {                                                                         \
            const float4* pw = (const float4*)((WROW) + kb + hf * 16);            \
            float4 w0 = pw[0], w1 = pw[1], w2 = pw[2], w3 = pw[3];                \
            v16bf bfr;                                                            \
            bfr[0]  = (__bf16)w0.x; bfr[1]  = (__bf16)w0.y;                       \
            bfr[2]  = (__bf16)w0.z; bfr[3]  = (__bf16)w0.w;                       \
            bfr[4]  = (__bf16)w1.x; bfr[5]  = (__bf16)w1.y;                       \
            bfr[6]  = (__bf16)w1.z; bfr[7]  = (__bf16)w1.w;                       \
            bfr[8]  = (__bf16)w2.x; bfr[9]  = (__bf16)w2.y;                       \
            bfr[10] = (__bf16)w2.z; bfr[11] = (__bf16)w2.w;                       \
            bfr[12] = (__bf16)w3.x; bfr[13] = (__bf16)w3.y;                       \
            bfr[14] = (__bf16)w3.z; bfr[15] = (__bf16)w3.w;                       \
            ACC = __builtin_amdgcn_wmma_f32_16x16x32_bf16(                        \
                false, afr, false, bfr, (short)0, ACC, false, false);             \
        }

        TILE_STEP(a0, wbase + (size_t)0 * 16 * Dsz);
        TILE_STEP(a1, wbase + (size_t)1 * 16 * Dsz);
        TILE_STEP(a2, wbase + (size_t)2 * 16 * Dsz);
        TILE_STEP(a3, wbase + (size_t)3 * 16 * Dsz);
        TILE_STEP(a4, wbase + (size_t)4 * 16 * Dsz);
        TILE_STEP(a5, wbase + (size_t)5 * 16 * Dsz);
        TILE_STEP(a6, wbase + (size_t)6 * 16 * Dsz);
        TILE_STEP(a7, wbase + (size_t)7 * 16 * Dsz);
        if (w >= 6) TILE_STEP(a8, wextra);
        #undef TILE_STEP

        __syncthreads();   // reads of slab[buf] done before it is overwritten
    }

    // Epilogue: C/D layout -> value r at (row = rt*16 + r + 8*hf, col = n)
    {
        auto store_delta = [&](const v8f& acc, int j) {
            int d = (w * 8 + j) * 16 + n;
            float bias = delta_b[d] + dt_bias[d];
            #pragma unroll
            for (int r = 0; r < 8; ++r) {
                int row = rt * 16 + r + 8 * hf;
                float p = acc[r] + bias;
                float sp = (p > 20.f) ? p : log1pf(__expf(p));
                delta[(size_t)row * Dsz + d] = sp;
            }
        };
        store_delta(a0, 0); store_delta(a1, 1); store_delta(a2, 2); store_delta(a3, 3);
        store_delta(a4, 4); store_delta(a5, 5); store_delta(a6, 6); store_delta(a7, 7);
        if (w == 6) {
            float bias = B_b[n];
            #pragma unroll
            for (int r = 0; r < 8; ++r)
                Bm[(size_t)(rt * 16 + r + 8 * hf) * Nsz + n] = a8[r] + bias;
        } else if (w == 7) {
            float bias = C_b[n];
            #pragma unroll
            for (int r = 0; r < 8; ++r)
                Cm[(size_t)(rt * 16 + r + 8 * hf) * Nsz + n] = a8[r] + bias;
        }
    }
}

// ---------------- Kernel 4: fused discretization + sequential scan ----------------
__global__ __launch_bounds__(256) void k_scan(
    const float* __restrict__ x, const float* __restrict__ sel,
    const float* __restrict__ delta, const float* __restrict__ Bm,
    const float* __restrict__ Cm, const float* __restrict__ A_log,
    float* __restrict__ y) {
    int tid = blockIdx.x * blockDim.x + threadIdx.x;
    int g = tid >> 4;
    int n = tid & 15;
    int b = g / Dsz, d = g % Dsz;
    float A = -__expf(A_log[(size_t)d * Nsz + n]);
    float sbd = sel[(size_t)b * Dsz + d];
    const float* dr = delta + ((size_t)b * Lsz) * Dsz + d;
    const float* xr = x     + ((size_t)b * Lsz) * Dsz + d;
    const float* br = Bm    + ((size_t)b * Lsz) * Nsz + n;
    const float* cr = Cm    + ((size_t)b * Lsz) * Nsz + n;
    float* yr       = y     + ((size_t)b * Lsz) * Dsz + d;
    bool smallA = fabsf(A) < 1e-6f;
    float invA = smallA ? 0.f : 1.f / A;
    float h = 0.f;
    for (int t = 0; t < Lsz; ++t) {
        float dlt = dr[(size_t)t * Dsz];
        float xv  = xr[(size_t)t * Dsz] * sbd;
        float bmv = br[(size_t)t * Nsz];
        float cmv = cr[(size_t)t * Nsz];
        float Ad = __expf(A * dlt);
        float factor = smallA ? dlt : (Ad - 1.f) * invA;
        h = Ad * h + factor * bmv * xv;
        float yv = h * cmv;
        yv += __shfl_xor(yv, 8, 32);
        yv += __shfl_xor(yv, 4, 32);
        yv += __shfl_xor(yv, 2, 32);
        yv += __shfl_xor(yv, 1, 32);
        if (n == 0) yr[(size_t)t * Dsz] = yv;
    }
}

extern "C" void kernel_launch(void* const* d_in, const int* in_sizes, int n_in,
                              void* d_out, int out_size, void* d_ws, size_t ws_size,
                              hipStream_t stream) {
    const float* x       = (const float*)d_in[0];
    const float* delta_W = (const float*)d_in[1];
    const float* delta_b = (const float*)d_in[2];
    const float* B_W     = (const float*)d_in[3];
    const float* B_b     = (const float*)d_in[4];
    const float* C_W     = (const float*)d_in[5];
    const float* C_b     = (const float*)d_in[6];
    const float* A_log   = (const float*)d_in[7];
    const float* dt_bias = (const float*)d_in[8];
    const float* tau     = (const float*)d_in[9];
    const float* Wi_W    = (const float*)d_in[10];
    const float* Wi_b    = (const float*)d_in[11];
    const float* Wr_W    = (const float*)d_in[12];
    const float* sel_W   = (const float*)d_in[13];
    const float* sel_b   = (const float*)d_in[14];
    const float* h0      = (const float*)d_in[15];
    float* out = (float*)d_out;

    float* wsf   = (float*)d_ws;
    float* delta = wsf;                                  // B*L*D
    float* Bm    = delta + (size_t)Bsz * Lsz * Dsz;      // B*L*N
    float* Cm    = Bm    + (size_t)Bsz * Lsz * Nsz;      // B*L*N
    float* gc    = Cm    + (size_t)Bsz * Lsz * Nsz;      // B*D
    float* sel   = gc    + (size_t)Bsz * Dsz;            // B*D

    k_gc<<<(Bsz * Dsz) / 32, 256, 0, stream>>>(x, gc);
    k_liquid<<<1, 256, 0, stream>>>(gc, Wi_W, Wi_b, Wr_W, sel_W, sel_b, tau, h0, sel);
    k_gemm<<<(Bsz * Lsz) / 16, 256, 0, stream>>>(x, sel, delta_W, delta_b, dt_bias,
                                                 B_W, B_b, C_W, C_b, delta, Bm, Cm);
    k_scan<<<(Bsz * Dsz * 16) / 256, 256, 0, stream>>>(x, sel, delta, Bm, Cm, A_log, out);
}
